// Transformer_68289980006846
// MI455X (gfx1250) — matrix operand order repaired and checked by reading, not compiled, not run on hardware
//
#include <hip/hip_runtime.h>
#include <hip/hip_bf16.h>
#include <math.h>

// ---------------------------------------------------------------------------
// CDNA5 (gfx1250) transformer forward.  All GEMMs run on v_wmma_f32_16x16x32_bf16
// (fp32 -> bf16 conversion at the LDS staging step, fp32 accumulate).
// 128x64x32 block tile, 8 waves, each wave owns a 32x32 output slab:
// 2 A-frags + 2 B-frags -> 4 WMMAs per K-step (every fragment reused twice).
// All GEMM shapes in this model are exact multiples of the tile, so the
// kernels carry NO bounds checks.  B-operand addressing (contiguous-K vs
// strided-K) is a compile-time template.  Epilogue uses pointer-increment
// addressing (no per-element 64-bit multiplies).
// ---------------------------------------------------------------------------

typedef __attribute__((ext_vector_type(8)))  __bf16 v8bf;
typedef __attribute__((ext_vector_type(16))) __bf16 v16bf;
typedef __attribute__((ext_vector_type(8)))  float  v8f;

#define BM 128
#define BN 64
#define BKK 32
#define KPAD 8          // LDS row stride = 40 halves = 80 bytes (16B-aligned chunks)

__device__ __forceinline__ v8bf cvt8(const float* __restrict__ s)
{
    const float4 f0 = *(const float4*)(s);
    const float4 f1 = *(const float4*)(s + 4);
    v8bf o = { (__bf16)f0.x, (__bf16)f0.y, (__bf16)f0.z, (__bf16)f0.w,
               (__bf16)f1.x, (__bf16)f1.y, (__bf16)f1.z, (__bf16)f1.w };
    return o;
}

__device__ __forceinline__ v16bf ldfrag(const __bf16* p0, const __bf16* p1)
{
    v8bf lo = *(const v8bf*)p0;
    v8bf hi = *(const v8bf*)p1;
    return __builtin_shufflevector(lo, hi, 0,1,2,3,4,5,6,7,8,9,10,11,12,13,14,15);
}

// Tiled WMMA GEMM (tile-aligned, no bounds checks):
//   C[m,n] = act( scale * sum_k A[m*lda + k] * B[n*ldbN + k*ldbK] + bias[n] ) + Res[m*ldc+n]
// Batched over blockIdx.z with (b,h) = (z/nh, z%nh) offsets for attention use.
template<bool BSTRIDED>
__global__ __launch_bounds__(256)
void gemm_wmma_bf16(const float* __restrict__ A, const float* __restrict__ Bm,
                    const float* __restrict__ bias, const float* __restrict__ Res,
                    float* __restrict__ C,
                    int K,
                    int lda, long ldbN, long ldbK, int ldc,
                    float scale, int act, int nh,
                    long sAb, long sAh, long sBb, long sBh, long sCb, long sCh)
{
    __shared__ __align__(16) __bf16 lsA[BM][BKK + KPAD];   // [m][k]
    __shared__ __align__(16) __bf16 lsB[BN][BKK + KPAD];   // [n][k]  (K-major per column)

    const int z  = blockIdx.z;
    const int bb = z / nh;
    const int hh = z - bb * nh;
    A  += (size_t)bb * sAb + (size_t)hh * sAh;
    Bm += (size_t)bb * sBb + (size_t)hh * sBh;
    C  += (size_t)bb * sCb + (size_t)hh * sCh;

    const int n0 = blockIdx.x * BN;
    const int m0 = blockIdx.y * BM;

    const int tid  = threadIdx.x;
    const int lane = tid & 31;
    const int wave = tid >> 5;     // 8 waves
    const int wm   = wave & 3;     // 4 row-slabs of 32
    const int wn   = wave >> 2;    // 2 col-slabs of 32

    const int hsel = lane >> 4;    // lane half-group (ISA 16-bit fragment layout)
    const int r16  = lane & 15;

    v8f acc00 = {}, acc01 = {}, acc10 = {}, acc11 = {};

    // A staging: 128 rows x 32 k = 4096 elems; 256 thr x 16 contiguous k each.
    const int arS = tid >> 1;            // 0..127
    const int akS = (tid & 1) * 16;      // 0 or 16
    // B staging: 64 rows x 32 k = 2048 elems; 256 thr x 8 contiguous k each.
    const int brS = tid >> 2;            // 0..63
    const int bkS = (tid & 3) * 8;       // 0,8,16,24

    const float* aPtr = A + (size_t)(m0 + arS) * lda + akS;
    const float* bPtr = Bm + (size_t)(n0 + brS) * ldbN + (BSTRIDED ? (size_t)bkS * ldbK
                                                                   : (size_t)bkS);
    __bf16* dstA = &lsA[arS][akS];
    __bf16* dstB = &lsB[brS][bkS];

    for (int k0 = 0; k0 < K; k0 += BKK) {
        __syncthreads();
        // ---- stage A tile (fp32 -> bf16): 16 contiguous floats per thread ----
        *(v8bf*)(dstA)     = cvt8(aPtr);
        *(v8bf*)(dstA + 8) = cvt8(aPtr + 8);
        aPtr += BKK;
        // ---- stage B tile as [n][k] ----
        if (!BSTRIDED) {
            *(v8bf*)dstB = cvt8(bPtr);
            bPtr += BKK;
        } else {
            v8bf o;
            #pragma unroll
            for (int i = 0; i < 8; ++i)
                o[i] = (__bf16)bPtr[(size_t)i * ldbK];
            *(v8bf*)dstB = o;
            bPtr += (size_t)BKK * ldbK;
        }
        __syncthreads();

        // A fragments (16x32 bf16): lanes 0-15 hold K {0..7,16..23}, lanes 16-31 K {8..15,24..31}
        const __bf16* ar0 = &lsA[wm * 32 + r16][0];
        const __bf16* ar1 = &lsA[wm * 32 + 16 + r16][0];
        v16bf a0 = ldfrag(ar0 + hsel * 8, ar0 + 16 + hsel * 8);
        v16bf a1 = ldfrag(ar1 + hsel * 8, ar1 + 16 + hsel * 8);

        // B fragments (32x16 bf16): lanes 0-15 hold K {0..7,16..23}, lanes 16-31 K {8..15,24..31}, col = lane&15
        const __bf16* br0 = &lsB[wn * 32 + r16][hsel * 8];
        const __bf16* br1 = &lsB[wn * 32 + 16 + r16][hsel * 8];
        v16bf b0 = ldfrag(br0, br0 + 16);
        v16bf b1 = ldfrag(br1, br1 + 16);

        acc00 = __builtin_amdgcn_wmma_f32_16x16x32_bf16(false, a0, false, b0,
                                                        (short)0, acc00, false, false);
        acc01 = __builtin_amdgcn_wmma_f32_16x16x32_bf16(false, a0, false, b1,
                                                        (short)0, acc01, false, false);
        acc10 = __builtin_amdgcn_wmma_f32_16x16x32_bf16(false, a1, false, b0,
                                                        (short)0, acc10, false, false);
        acc11 = __builtin_amdgcn_wmma_f32_16x16x32_bf16(false, a1, false, b1,
                                                        (short)0, acc11, false, false);
    }

    // Epilogue: C fragment layout — VGPR r: lanes 0-15 M=r, lanes 16-31 M=8+r; N = lane&15.
    // Each lane's 8 stores walk consecutive rows at a fixed column: use one base
    // address per sub-tile and bump by ldc (no per-element 64-bit multiplies).
    #pragma unroll
    for (int tm = 0; tm < 2; ++tm) {
        #pragma unroll
        for (int tn = 0; tn < 2; ++tn) {
            v8f a = tm ? (tn ? acc11 : acc10) : (tn ? acc01 : acc00);
            const int col  = n0 + wn * 32 + tn * 16 + r16;
            const int row0 = m0 + wm * 32 + tm * 16 + hsel * 8;
            const float bv = bias ? bias[col] : 0.0f;
            float*       cp = C + (size_t)row0 * ldc + col;
            const float* rp = Res ? (Res + (size_t)row0 * ldc + col) : cp;
            #pragma unroll
            for (int r = 0; r < 8; ++r) {
                float v = a[r] * scale + bv;
                if (act == 1) v = fmaxf(v, 0.0f);
                if (Res) v += rp[0];
                cp[0] = v;
                cp += ldc;
                rp += ldc;
            }
        }
    }
}

// Masked softmax over the last dim of scores[b][h][q][k].
__global__ __launch_bounds__(256)
void softmax_mask(float* __restrict__ S, const int* __restrict__ keys,
                  int nh, int Lq, int Lk, int causal)
{
    __shared__ float red[256];
    const int row = blockIdx.x;                 // (b*nh + h)*Lq + q
    const int q   = row % Lq;
    const int b   = row / (nh * Lq);
    float* s = S + (size_t)row * Lk;
    const int* kt = keys + (size_t)b * Lk;
    const int tid = threadIdx.x;

    float mx = -3.4e38f;
    for (int k = tid; k < Lk; k += 256) {
        float v = s[k];
        const bool m = (kt[k] == 0) || (causal && k > q);
        if (m) v = -1.0e9f;
        s[k] = v;
        mx = fmaxf(mx, v);
    }
    red[tid] = mx; __syncthreads();
    for (int off = 128; off > 0; off >>= 1) {
        if (tid < off) red[tid] = fmaxf(red[tid], red[tid + off]);
        __syncthreads();
    }
    mx = red[0]; __syncthreads();

    float sum = 0.0f;
    for (int k = tid; k < Lk; k += 256) {
        const float e = __expf(s[k] - mx);
        s[k] = e;
        sum += e;
    }
    red[tid] = sum; __syncthreads();
    for (int off = 128; off > 0; off >>= 1) {
        if (tid < off) red[tid] += red[tid + off];
        __syncthreads();
    }
    const float inv = 1.0f / red[0];
    for (int k = tid; k < Lk; k += 256) s[k] *= inv;
}

// Row-wise LayerNorm over Dm features.
__global__ __launch_bounds__(128)
void layernorm_k(const float* __restrict__ X, const float* __restrict__ g,
                 const float* __restrict__ bta, float* __restrict__ Y, int Dm)
{
    __shared__ float red[128];
    const int row = blockIdx.x;
    const float* x = X + (size_t)row * Dm;
    float* y = Y + (size_t)row * Dm;
    const int tid = threadIdx.x;

    float s = 0.0f;
    for (int d = tid; d < Dm; d += 128) s += x[d];
    red[tid] = s; __syncthreads();
    for (int off = 64; off > 0; off >>= 1) {
        if (tid < off) red[tid] += red[tid + off];
        __syncthreads();
    }
    const float mean = red[0] / (float)Dm; __syncthreads();

    float s2 = 0.0f;
    for (int d = tid; d < Dm; d += 128) { const float t = x[d] - mean; s2 += t * t; }
    red[tid] = s2; __syncthreads();
    for (int off = 64; off > 0; off >>= 1) {
        if (tid < off) red[tid] += red[tid + off];
        __syncthreads();
    }
    const float rstd = rsqrtf(red[0] / (float)Dm + 1e-5f);
    for (int d = tid; d < Dm; d += 128) y[d] = (x[d] - mean) * rstd * g[d] + bta[d];
}

__global__ void embed_k(const int* __restrict__ tok, const float* __restrict__ emb,
                        const float* __restrict__ pe, float* __restrict__ out,
                        int T, int Lseq, int Dm, float scale)
{
    const int i = blockIdx.x * blockDim.x + threadIdx.x;
    if (i >= T * Dm) return;
    const int t = i / Dm, d = i - t * Dm;
    const int pos = t % Lseq;
    out[i] = emb[(size_t)tok[t] * Dm + d] * scale + pe[(size_t)pos * Dm + d];
}

__global__ void add_k(float* __restrict__ y, const float* __restrict__ a, int n)
{
    const int i = blockIdx.x * blockDim.x + threadIdx.x;
    if (i < n) y[i] += a[i];
}

// ---------------------------------------------------------------------------

static inline void launch_gemm(hipStream_t st, bool bstrided,
    const float* A, const float* Bm, const float* bias, const float* Res, float* C,
    int M, int N, int K, int lda, long ldbN, long ldbK, int ldc,
    float scale, int act,
    int batches, int nh,
    long sAb, long sAh, long sBb, long sBh, long sCb, long sCh)
{
    dim3 grid((N + BN - 1) / BN, (M + BM - 1) / BM, batches);
    if (!bstrided)
        gemm_wmma_bf16<false><<<grid, 256, 0, st>>>(A, Bm, bias, Res, C, K,
                                                    lda, ldbN, ldbK, ldc, scale, act, nh,
                                                    sAb, sAh, sBb, sBh, sCb, sCh);
    else
        gemm_wmma_bf16<true><<<grid, 256, 0, st>>>(A, Bm, bias, Res, C, K,
                                                   lda, ldbN, ldbK, ldc, scale, act, nh,
                                                   sAb, sAh, sBb, sBh, sCb, sCh);
}

extern "C" void kernel_launch(void* const* d_in, const int* in_sizes, int n_in,
                              void* d_out, int out_size, void* d_ws, size_t ws_size,
                              hipStream_t stream)
{
    (void)in_sizes; (void)n_in; (void)out_size; (void)ws_size;

    constexpr int Bn = 4, LS = 512, LT = 512, Dm = 512, DFF = 2048;
    constexpr int Lly = 3, NH = 8, DK = 64, VD = 32000;
    constexpr int T = Bn * LS;                  // 2048 tokens (both streams)
    const float embScale = 22.62741699796952f;  // sqrt(512)
    const float qkScale  = 0.125f;              // 1/sqrt(64)

    // ---- inputs (setup_inputs order) ----
    const int*   enc_inputs = (const int*)  d_in[0];
    const int*   dec_inputs = (const int*)  d_in[1];
    const float* enc_emb    = (const float*)d_in[2];
    const float* dec_emb    = (const float*)d_in[3];
    const float* pe         = (const float*)d_in[4];
    const float* eWq  = (const float*)d_in[5];  const float* eWqb  = (const float*)d_in[6];
    const float* eWk  = (const float*)d_in[7];  const float* eWkb  = (const float*)d_in[8];
    const float* eWv  = (const float*)d_in[9];  const float* eWvb  = (const float*)d_in[10];
    const float* eWo  = (const float*)d_in[11]; const float* eWob  = (const float*)d_in[12];
    const float* dsWq = (const float*)d_in[13]; const float* dsWqb = (const float*)d_in[14];
    const float* dsWk = (const float*)d_in[15]; const float* dsWkb = (const float*)d_in[16];
    const float* dsWv = (const float*)d_in[17]; const float* dsWvb = (const float*)d_in[18];
    const float* dsWo = (const float*)d_in[19]; const float* dsWob = (const float*)d_in[20];
    const float* dcWq = (const float*)d_in[21]; const float* dcWqb = (const float*)d_in[22];
    const float* dcWk = (const float*)d_in[23]; const float* dcWkb = (const float*)d_in[24];
    const float* dcWv = (const float*)d_in[25]; const float* dcWvb = (const float*)d_in[26];
    const float* dcWo = (const float*)d_in[27]; const float* dcWob = (const float*)d_in[28];
    const float* eF1  = (const float*)d_in[29]; const float* eF1b  = (const float*)d_in[30];
    const float* eF2  = (const float*)d_in[31]; const float* eF2b  = (const float*)d_in[32];
    const float* dF1  = (const float*)d_in[33]; const float* dF1b  = (const float*)d_in[34];
    const float* dF2  = (const float*)d_in[35]; const float* dF2b  = (const float*)d_in[36];
    const float* eN1g = (const float*)d_in[37]; const float* eN1b  = (const float*)d_in[38];
    const float* eN2g = (const float*)d_in[39]; const float* eN2b  = (const float*)d_in[40];
    const float* dN1g = (const float*)d_in[41]; const float* dN1b  = (const float*)d_in[42];
    const float* dN2g = (const float*)d_in[43]; const float* dN2b  = (const float*)d_in[44];
    const float* dN3g = (const float*)d_in[45]; const float* dN3b  = (const float*)d_in[46];
    const float* eNfg = (const float*)d_in[47]; const float* eNfb  = (const float*)d_in[48];
    const float* projW = (const float*)d_in[49]; const float* projb = (const float*)d_in[50];

    float* out = (float*)d_out;

    // ---- workspace layout (floats) ----
    const size_t TD = (size_t)T * Dm;            // 1,048,576
    float* ws  = (float*)d_ws;
    float* x    = ws + 0 * TD;
    float* h    = ws + 1 * TD;
    float* q    = ws + 2 * TD;
    float* kb   = ws + 3 * TD;
    float* v    = ws + 4 * TD;
    float* ctx  = ws + 5 * TD;
    float* enc  = ws + 6 * TD;
    float* y    = ws + 7 * TD;
    float* ca   = ws + 8 * TD;
    float* ffn  = ws + 9 * TD;                   // T*DFF = 4*TD  -> [9,13)
    float* sc   = ws + 13 * TD;                  // B*NH*512*512 = 8*TD -> [13,21)

    const long sQKb = (long)LS * Dm;             // per-batch stride into token-major act
    const long sHd  = (long)DK;                  // per-head column offset
    const long sScb = (long)NH * LS * LS;        // scores per-batch
    const long sSch = (long)LS * LS;             // scores per-head

    // ======================= Encoder =======================
    embed_k<<<(T * Dm + 255) / 256, 256, 0, stream>>>(enc_inputs, enc_emb, pe, x,
                                                      T, LS, Dm, embScale);
    for (int l = 0; l < Lly; ++l) {
        const size_t wo = (size_t)l * Dm * Dm;
        layernorm_k<<<T, 128, 0, stream>>>(x, eN1g + l * Dm, eN1b + l * Dm, h, Dm);
        launch_gemm(stream, false, h, eWq + wo, eWqb + l * Dm, nullptr, q,  T, Dm, Dm, Dm, Dm, 1, Dm, 1.f, 0, 1,1,0,0,0,0,0,0);
        launch_gemm(stream, false, h, eWk + wo, eWkb + l * Dm, nullptr, kb, T, Dm, Dm, Dm, Dm, 1, Dm, 1.f, 0, 1,1,0,0,0,0,0,0);
        launch_gemm(stream, false, h, eWv + wo, eWvb + l * Dm, nullptr, v,  T, Dm, Dm, Dm, Dm, 1, Dm, 1.f, 0, 1,1,0,0,0,0,0,0);
        // scores = Q K^T / sqrt(dk)   (batched over b,h)
        launch_gemm(stream, false, q, kb, nullptr, nullptr, sc,
                    LS, LS, DK, Dm, (long)Dm, 1, LS, qkScale, 0,
                    Bn * NH, NH, sQKb, sHd, sQKb, sHd, sScb, sSch);
        softmax_mask<<<Bn * NH * LS, 256, 0, stream>>>(sc, enc_inputs, NH, LS, LS, 0);
        // ctx = attn @ V   (B strided in K)
        launch_gemm(stream, true, sc, v, nullptr, nullptr, ctx,
                    LS, DK, LS, LS, 1, (long)Dm, Dm, 1.f, 0,
                    Bn * NH, NH, sScb, sSch, sQKb, sHd, sQKb, sHd);
        // x = x + ctx @ Wo^T + bo
        launch_gemm(stream, false, ctx, eWo + wo, eWob + l * Dm, x, x, T, Dm, Dm, Dm, Dm, 1, Dm, 1.f, 0, 1,1,0,0,0,0,0,0);
        layernorm_k<<<T, 128, 0, stream>>>(x, eN2g + l * Dm, eN2b + l * Dm, h, Dm);
        launch_gemm(stream, false, h,   eF1 + (size_t)l * DFF * Dm, eF1b + l * DFF, nullptr, ffn,
                    T, DFF, Dm, Dm, Dm, 1, DFF, 1.f, 1, 1,1,0,0,0,0,0,0);
        launch_gemm(stream, false, ffn, eF2 + (size_t)l * Dm * DFF, eF2b + l * Dm,  x, x,
                    T, Dm, DFF, DFF, DFF, 1, Dm, 1.f, 0, 1,1,0,0,0,0,0,0);
    }
    layernorm_k<<<T, 128, 0, stream>>>(x, eNfg, eNfb, enc, Dm);

    // ======================= Decoder =======================
    embed_k<<<(T * Dm + 255) / 256, 256, 0, stream>>>(dec_inputs, dec_emb, pe, y,
                                                      T, LT, Dm, embScale);
    for (int l = 0; l < Lly; ++l) {
        const size_t wo = (size_t)l * Dm * Dm;
        // ---- masked self-attention ----
        layernorm_k<<<T, 128, 0, stream>>>(y, dN1g + l * Dm, dN1b + l * Dm, h, Dm);
        launch_gemm(stream, false, h, dsWq + wo, dsWqb + l * Dm, nullptr, q,  T, Dm, Dm, Dm, Dm, 1, Dm, 1.f, 0, 1,1,0,0,0,0,0,0);
        launch_gemm(stream, false, h, dsWk + wo, dsWkb + l * Dm, nullptr, kb, T, Dm, Dm, Dm, Dm, 1, Dm, 1.f, 0, 1,1,0,0,0,0,0,0);
        launch_gemm(stream, false, h, dsWv + wo, dsWvb + l * Dm, nullptr, v,  T, Dm, Dm, Dm, Dm, 1, Dm, 1.f, 0, 1,1,0,0,0,0,0,0);
        launch_gemm(stream, false, q, kb, nullptr, nullptr, sc,
                    LT, LT, DK, Dm, (long)Dm, 1, LT, qkScale, 0,
                    Bn * NH, NH, sQKb, sHd, sQKb, sHd, sScb, sSch);
        softmax_mask<<<Bn * NH * LT, 256, 0, stream>>>(sc, dec_inputs, NH, LT, LT, 1);
        launch_gemm(stream, true, sc, v, nullptr, nullptr, ctx,
                    LT, DK, LT, LT, 1, (long)Dm, Dm, 1.f, 0,
                    Bn * NH, NH, sScb, sSch, sQKb, sHd, sQKb, sHd);
        launch_gemm(stream, false, ctx, dsWo + wo, dsWob + l * Dm, y, y, T, Dm, Dm, Dm, Dm, 1, Dm, 1.f, 0, 1,1,0,0,0,0,0,0);
        // ---- cross-attention ----
        layernorm_k<<<T, 128, 0, stream>>>(y, dN2g + l * Dm, dN2b + l * Dm, h, Dm);
        launch_gemm(stream, false, h,   dcWq + wo, dcWqb + l * Dm, nullptr, q,  T, Dm, Dm, Dm, Dm, 1, Dm, 1.f, 0, 1,1,0,0,0,0,0,0);
        launch_gemm(stream, false, enc, dcWk + wo, dcWkb + l * Dm, nullptr, kb, T, Dm, Dm, Dm, Dm, 1, Dm, 1.f, 0, 1,1,0,0,0,0,0,0);
        launch_gemm(stream, false, enc, dcWv + wo, dcWvb + l * Dm, nullptr, v,  T, Dm, Dm, Dm, Dm, 1, Dm, 1.f, 0, 1,1,0,0,0,0,0,0);
        launch_gemm(stream, false, q, kb, nullptr, nullptr, sc,
                    LT, LS, DK, Dm, (long)Dm, 1, LS, qkScale, 0,
                    Bn * NH, NH, sQKb, sHd, sQKb, sHd, sScb, sSch);
        softmax_mask<<<Bn * NH * LT, 256, 0, stream>>>(sc, enc_inputs, NH, LT, LS, 0);
        launch_gemm(stream, true, sc, v, nullptr, nullptr, ctx,
                    LT, DK, LS, LS, 1, (long)Dm, Dm, 1.f, 0,
                    Bn * NH, NH, sScb, sSch, sQKb, sHd, sQKb, sHd);
        // ca = ctx @ Wo^T + bo ; y = y + ca
        launch_gemm(stream, false, ctx, dcWo + wo, dcWob + l * Dm, nullptr, ca, T, Dm, Dm, Dm, Dm, 1, Dm, 1.f, 0, 1,1,0,0,0,0,0,0);
        add_k<<<((int)TD + 255) / 256, 256, 0, stream>>>(y, ca, (int)TD);
        // ---- FFN with residual = ca ----
        layernorm_k<<<T, 128, 0, stream>>>(y, dN3g + l * Dm, dN3b + l * Dm, h, Dm);
        launch_gemm(stream, false, h,   dF1 + (size_t)l * DFF * Dm, dF1b + l * DFF, nullptr, ffn,
                    T, DFF, Dm, Dm, Dm, 1, DFF, 1.f, 1, 1,1,0,0,0,0,0,0);
        launch_gemm(stream, false, ffn, dF2 + (size_t)l * Dm * DFF, dF2b + l * Dm,  ca, y,
                    T, Dm, DFF, DFF, DFF, 1, Dm, 1.f, 0, 1,1,0,0,0,0,0,0);
    }

    // ---- final projection: logits = y @ projW^T + projb ----
    launch_gemm(stream, false, y, projW, projb, nullptr, out,
                T, VD, Dm, Dm, Dm, 1, VD, 1.f, 0, 1,1,0,0,0,0,0,0);
}
